// SpatiotemporalAttention_67851893342906
// MI455X (gfx1250) — compile-verified, matrix operand order checked by reading, not run on hardware
//
#include <hip/hip_runtime.h>
#include <hip/hip_bf16.h>

typedef _Float16 v8h  __attribute__((ext_vector_type(8)));
typedef _Float16 v16h __attribute__((ext_vector_type(16)));
typedef float    v8f  __attribute__((ext_vector_type(8)));

#define NIMG   128      // B*T
#define C64    64
#define HW     3136     // 56*56
#define HW4    784      // HW/4
#define TILES  196      // HW / 16
#define SEGS   4        // image split into 4 segments of 49 tiles
#define TSEG   49
#define LN_EPS 1e-5f

// -------- WMMA helper: D = A(16x32 f16) * B(32x16 f16) + C(f32) ----------
__device__ __forceinline__ v8f wmma32f16(v16h a, v16h b, v8f c) {
    return __builtin_amdgcn_wmma_f32_16x16x32_f16(
        /*neg_a=*/false, a, /*neg_b=*/false, b,
        /*c_mod=*/(short)0, c, /*reuse_a=*/false, /*reuse_b=*/false);
}

// A fragment: lane l holds row M = l%16; K chunks [ks*32+o8 .. +7] and [+16 .. +23]
// (o8 = 8 for upper half-wave). Two 16B LDS reads, concatenated.
__device__ __forceinline__ v16h load_a_frag(const _Float16* wrow, int ks, int o8) {
    const v8h* p0 = (const v8h*)(wrow + ks * 32 + o8);
    const v8h* p1 = (const v8h*)(wrow + ks * 32 + o8 + 16);
    v8h lo = *p0;
    v8h hi = *p1;
    return __builtin_shufflevector(lo, hi, 0,1,2,3,4,5,6,7,8,9,10,11,12,13,14,15);
}

// ---------------- kernel 0: zero pool accumulator -------------------------
__global__ void sta_kzero(float* __restrict__ p, int n) {
    int i = blockIdx.x * blockDim.x + threadIdx.x;
    if (i < n) p[i] = 0.0f;
}

// ---------------- kernel 1: fused conv1x1 chain + LN + pool ---------------
// grid = (128 images, 4 segments), block = 256 (8 waves)
__global__ __launch_bounds__(256) void sta_k1(
    const float* __restrict__ x,
    const float* __restrict__ w1, const float* __restrict__ b1,
    const float* __restrict__ w2, const float* __restrict__ b2,
    const float* __restrict__ lng, const float* __restrict__ lnb,
    const float* __restrict__ w3, const float* __restrict__ b3,
    float* __restrict__ saf_out,       // d_out: saf = x + gf
    float* __restrict__ gpsum)         // [128][64] pool accumulator (f32)
{
    __shared__ __align__(16) _Float16 sW[3][C64][C64];  // f16 weights, row-major [o][k]
    __shared__ float sB[3][C64];
    __shared__ float sG[C64], sBt[C64];

    const int tid = threadIdx.x;
    {
        _Float16* sw = &sW[0][0][0];
        for (int i = tid; i < C64 * C64; i += 256) {
            sw[i]        = (_Float16)w1[i];
            sw[4096 + i] = (_Float16)w2[i];
            sw[8192 + i] = (_Float16)w3[i];
        }
        if (tid < C64) {
            sB[0][tid] = b1[tid];
            sB[1][tid] = b2[tid];
            sB[2][tid] = b3[tid];
            sG[tid]  = lng[tid];
            sBt[tid] = lnb[tid];
        }
    }
    __syncthreads();

    const int lane = tid & 31;
    const int wid  = tid >> 5;
    const int n    = blockIdx.x;          // image index
    const int seg  = blockIdx.y;          // tile segment
    const int o8   = (lane & 16) ? 8 : 0; // channel offset of this half-wave
    const int prow = lane & 15;           // pixel-in-tile = lane%16

    const float* xn = x + (size_t)n * C64 * HW;
    float*       on = saf_out + (size_t)n * C64 * HW;

    float gacc[4][8];
    #pragma unroll
    for (int m = 0; m < 4; ++m)
        #pragma unroll
        for (int r = 0; r < 8; ++r) gacc[m][r] = 0.0f;

    const int t0 = seg * TSEG;
    for (int t = t0 + wid; t < t0 + TSEG; t += 8) {
        const int p = t * 16 + prow;

        // ---- load x tile: lane holds pixel p, channels {m*16+r+o8} --------
        float xe[4][8];
        #pragma unroll
        for (int m = 0; m < 4; ++m)
            #pragma unroll
            for (int r = 0; r < 8; ++r)
                xe[m][r] = xn[(size_t)(m * 16 + r + o8) * HW + p];

        // ---- pack B operand for GEMM1 (B mirrors A layout, lane-local) ----
        v16h B1[2];
        #pragma unroll
        for (int ks = 0; ks < 2; ++ks)
            #pragma unroll
            for (int e = 0; e < 8; ++e) {
                B1[ks][e]     = (_Float16)xe[2 * ks][e];
                B1[ks][8 + e] = (_Float16)xe[2 * ks + 1][e];
            }

        // ---- GEMM1: am = W1*x + b1 ; gate t = am*x ; pack B2 --------------
        v16h B2[2];
        #pragma unroll
        for (int mt = 0; mt < 4; ++mt) {
            const _Float16* wrow = &sW[0][mt * 16 + prow][0];
            v8f d = {};
            d = wmma32f16(load_a_frag(wrow, 0, o8), B1[0], d);
            d = wmma32f16(load_a_frag(wrow, 1, o8), B1[1], d);
            #pragma unroll
            for (int r = 0; r < 8; ++r) {
                float v = (d[r] + sB[0][mt * 16 + o8 + r]) * xe[mt][r];
                B2[mt >> 1][(mt & 1) * 8 + r] = (_Float16)v;
            }
        }

        // ---- GEMM2: gi = W2*t + b2 ; LayerNorm over 64 channels -----------
        float gi[4][8];
        float s1 = 0.0f, s2 = 0.0f;
        #pragma unroll
        for (int mt = 0; mt < 4; ++mt) {
            const _Float16* wrow = &sW[1][mt * 16 + prow][0];
            v8f d = {};
            d = wmma32f16(load_a_frag(wrow, 0, o8), B2[0], d);
            d = wmma32f16(load_a_frag(wrow, 1, o8), B2[1], d);
            #pragma unroll
            for (int r = 0; r < 8; ++r) {
                float v = d[r] + sB[1][mt * 16 + o8 + r];
                gi[mt][r] = v;
                s1 += v;
                s2 += v * v;
            }
        }
        // pixel's 64 channels live in lane pair (l, l^16)
        s1 += __shfl_xor(s1, 16);
        s2 += __shfl_xor(s2, 16);
        const float mean = s1 * (1.0f / 64.0f);
        const float var  = s2 * (1.0f / 64.0f) - mean * mean;
        const float rs   = rsqrtf(var + LN_EPS);

        // ---- gamma/beta + ReLU ; pack B3 ---------------------------------
        v16h B3[2];
        #pragma unroll
        for (int mt = 0; mt < 4; ++mt)
            #pragma unroll
            for (int r = 0; r < 8; ++r) {
                const int c = mt * 16 + o8 + r;
                float v = (gi[mt][r] - mean) * rs * sG[c] + sBt[c];
                v = fmaxf(v, 0.0f);
                B3[mt >> 1][(mt & 1) * 8 + r] = (_Float16)v;
            }

        // ---- GEMM3: gf = W3*relu + b3 ; saf = x + gf ; pool accum ---------
        #pragma unroll
        for (int mt = 0; mt < 4; ++mt) {
            const _Float16* wrow = &sW[2][mt * 16 + prow][0];
            v8f d = {};
            d = wmma32f16(load_a_frag(wrow, 0, o8), B3[0], d);
            d = wmma32f16(load_a_frag(wrow, 1, o8), B3[1], d);
            #pragma unroll
            for (int r = 0; r < 8; ++r) {
                const int c = mt * 16 + o8 + r;
                float gf = d[r] + sB[2][c];
                gacc[mt][r] += gf;
                on[(size_t)c * HW + p] = xe[mt][r] + gf;
            }
        }
    }

    // ---- reduce pool partials over the 16 pixels (lanes) of each half ----
    #pragma unroll
    for (int off = 1; off < 16; off <<= 1)
        #pragma unroll
        for (int m = 0; m < 4; ++m)
            #pragma unroll
            for (int r = 0; r < 8; ++r)
                gacc[m][r] += __shfl_xor(gacc[m][r], off);

    if ((lane & 15) == 0) {
        #pragma unroll
        for (int m = 0; m < 4; ++m)
            #pragma unroll
            for (int r = 0; r < 8; ++r)
                atomicAdd(&gpsum[n * C64 + m * 16 + o8 + r], gacc[m][r]);
    }
}

// ---------------- kernel 2: q/k/v + 16x16 attention -> tf -----------------
__global__ __launch_bounds__(256) void sta_k2(
    const float* __restrict__ gpsum,
    const float* __restrict__ qw, const float* __restrict__ qb,
    const float* __restrict__ kw, const float* __restrict__ kb,
    const float* __restrict__ vw, const float* __restrict__ vb,
    float* __restrict__ tf)
{
    __shared__ float sgp[NIMG * C64];
    __shared__ float sq[NIMG * C64];
    __shared__ float sk[NIMG * C64];
    __shared__ float sv[NIMG * C64];

    const int tid = threadIdx.x;
    for (int i = tid; i < NIMG * C64; i += 256)
        sgp[i] = gpsum[i] * (1.0f / (float)HW);
    __syncthreads();

    for (int i = tid; i < NIMG * C64; i += 256) {
        const int nI = i >> 6, c = i & 63;
        const float* g = &sgp[nI * C64];
        float aq = qb[c], ak = kb[c], av = vb[c];
        #pragma unroll 4
        for (int j = 0; j < C64; ++j) {
            const float gv = g[j];
            aq += gv * qw[c * C64 + j];
            ak += gv * kw[c * C64 + j];
            av += gv * vw[c * C64 + j];
        }
        sq[i] = aq; sk[i] = ak; sv[i] = av;
    }
    __syncthreads();

    if (tid < NIMG) {
        const int b = tid >> 4;
        const float* qv = &sq[tid * C64];
        float sc[16];
        float mx = -1e30f;
        #pragma unroll
        for (int s = 0; s < 16; ++s) {
            const float* kv = &sk[(b * 16 + s) * C64];
            float d = 0.0f;
            for (int j = 0; j < C64; ++j) d += qv[j] * kv[j];
            d *= 0.125f;                     // 1/sqrt(64)
            sc[s] = d;
            mx = fmaxf(mx, d);
        }
        float den = 0.0f;
        #pragma unroll
        for (int s = 0; s < 16; ++s) { sc[s] = __expf(sc[s] - mx); den += sc[s]; }
        const float inv = 1.0f / den;
        for (int c = 0; c < C64; ++c) {
            float acc = 0.0f;
            #pragma unroll
            for (int s = 0; s < 16; ++s) acc += sc[s] * sv[(b * 16 + s) * C64 + c];
            tf[tid * C64 + c] = acc * inv;
        }
    }
}

// ---------------- kernel 3: out = saf * tf + x (float4) -------------------
// grid = 128*64 blocks; block b owns row (n,c): 784 float4 elements.
// tf scalar loaded once per block; no per-element integer division.
__global__ __launch_bounds__(256) void sta_k3(
    const float* __restrict__ x,
    const float* __restrict__ tf,
    float* out)
{
    const int row = blockIdx.x;                  // n*64 + c
    const float tv = tf[row];
    const float4* xr = (const float4*)x + (size_t)row * HW4;
    float4*       orow = (float4*)out + (size_t)row * HW4;

    for (int i = threadIdx.x; i < HW4; i += 256) {
        const float4 s  = orow[i];
        const float4 xv = xr[i];
        float4 r;
        r.x = fmaf(s.x, tv, xv.x);
        r.y = fmaf(s.y, tv, xv.y);
        r.z = fmaf(s.z, tv, xv.z);
        r.w = fmaf(s.w, tv, xv.w);
        orow[i] = r;
    }
}

extern "C" void kernel_launch(void* const* d_in, const int* in_sizes, int n_in,
                              void* d_out, int out_size, void* d_ws, size_t ws_size,
                              hipStream_t stream) {
    const float* x   = (const float*)d_in[0];
    const float* w1  = (const float*)d_in[1];
    const float* b1  = (const float*)d_in[2];
    const float* w2  = (const float*)d_in[3];
    const float* b2  = (const float*)d_in[4];
    const float* lng = (const float*)d_in[5];
    const float* lnb = (const float*)d_in[6];
    const float* w3  = (const float*)d_in[7];
    const float* b3  = (const float*)d_in[8];
    const float* qw  = (const float*)d_in[9];
    const float* qb  = (const float*)d_in[10];
    const float* kw  = (const float*)d_in[11];
    const float* kb  = (const float*)d_in[12];
    const float* vw  = (const float*)d_in[13];
    const float* vb  = (const float*)d_in[14];

    float* out   = (float*)d_out;
    float* gpsum = (float*)d_ws;               // 128*64 floats
    float* tf    = gpsum + NIMG * C64;         // 128*64 floats

    sta_kzero<<<(NIMG * C64 + 255) / 256, 256, 0, stream>>>(gpsum, NIMG * C64);

    sta_k1<<<dim3(NIMG, SEGS), 256, 0, stream>>>(
        x, w1, b1, w2, b2, lng, lnb, w3, b3, out, gpsum);

    sta_k2<<<1, 256, 0, stream>>>(gpsum, qw, qb, kw, kb, vw, vb, tf);

    sta_k3<<<NIMG * C64, 256, 0, stream>>>(x, tf, out);
}
